// WeightedSAGEConv_23278722744483
// MI455X (gfx1250) — compile-verified, matrix operand order checked by reading, not run on hardware
//
#include <hip/hip_runtime.h>
#include <hip/hip_bf16.h>

// ---------------------------------------------------------------------------
// WeightedSAGEConv forward for MI455X (gfx1250, wave32, WMMA).
//   n_src = relu(h_src @ Q_w^T + Q_b)                  [bf16 WMMA, f32 accum]
//   accum[d] += w_e * n_src[s];  wsum[d] += w_e        [global_atomic_add_f32]
//   z = relu(concat(accum/max(wsum,1), h_dst) @ W_w^T + W_b)   [bf16 WMMA]
// ---------------------------------------------------------------------------

#define D_IN  128
#define D_HID 128
#define D_K2  256   // D_HID + D_IN
#define N_SRC 50000
#define N_DST 50000

typedef __attribute__((ext_vector_type(16))) __bf16 v16bf;
typedef __attribute__((ext_vector_type(8)))  float  v8f;

// Operand vector viewed either as 16 bf16 or 8 packed dwords.
union V16 {
    v16bf     bf;
    unsigned  u[8];
};

// Pack two fp32 into one dword of two bf16 (truncation: bf16 = top 16 bits).
// Single v_perm_b32: dst = {hi[31:16], lo[31:16]}  (selector bytes 7,6,3,2).
__device__ __forceinline__ unsigned pk2(float lo, float hi) {
    union { float f; unsigned u; } a, b;
    a.f = lo; b.f = hi;
    return __builtin_amdgcn_perm(b.u, a.u, 0x07060302u);
}

__device__ __forceinline__ void pack8(V16& dst, int dw, float4 lo, float4 hi) {
    dst.u[dw + 0] = pk2(lo.x, lo.y);
    dst.u[dw + 1] = pk2(lo.z, lo.w);
    dst.u[dw + 2] = pk2(hi.x, hi.y);
    dst.u[dw + 3] = pk2(hi.z, hi.w);
}

// ---------------------------------------------------------------------------
// Zero scratch (accum + wsum), grid-stride float4 stores.
// ---------------------------------------------------------------------------
__global__ void __launch_bounds__(256)
zero_f4(float4* __restrict__ p, long n4) {
    long i = (long)blockIdx.x * blockDim.x + threadIdx.x;
    long stride = (long)gridDim.x * blockDim.x;
    float4 z = make_float4(0.f, 0.f, 0.f, 0.f);
    for (; i < n4; i += stride) p[i] = z;
}

// ---------------------------------------------------------------------------
// GEMM1: out[m,n] = relu( sum_k h[m,k] * Qw[n,k] + Qb[n] ),  M x 128 x 128.
// One wave per 16x16 C tile; block = 8 waves = one 16-row strip (all 8 N tiles).
// A layout (16-bit A 16x32, ISA 7.12.2): lane = (khalf<<4)|m,
//   dwords 0..3 -> K = k0 + khalf*8 + [0..7], dwords 4..7 -> K = +16.
// B = Qw rows (Qw is [N,K] row-major == B^T), same per-lane striping, n=lane&15.
// ---------------------------------------------------------------------------
__global__ void __launch_bounds__(256)
gemm1_relu(const float* __restrict__ h, const float* __restrict__ Qw,
           const float* __restrict__ Qb, float* __restrict__ out) {
    const int wave  = threadIdx.x >> 5;     // 0..7 -> N tile
    const int lane  = threadIdx.x & 31;
    const int mt    = blockIdx.x;           // M tile
    const int r16   = lane & 15;
    const int khalf = lane >> 4;

    const float* Arow = h  + (size_t)(mt * 16 + r16) * D_IN;
    const float* Brow = Qw + (size_t)(wave * 16 + r16) * D_IN;

    v8f c = {};
#pragma unroll
    for (int k0 = 0; k0 < D_IN; k0 += 32) {
        const float4* A4a = (const float4*)(Arow + k0 + khalf * 8);
        const float4* A4b = (const float4*)(Arow + k0 + 16 + khalf * 8);
        const float4* B4a = (const float4*)(Brow + k0 + khalf * 8);
        const float4* B4b = (const float4*)(Brow + k0 + 16 + khalf * 8);
        V16 a, b;
        pack8(a, 0, A4a[0], A4a[1]);
        pack8(a, 4, A4b[0], A4b[1]);
        pack8(b, 0, B4a[0], B4a[1]);
        pack8(b, 4, B4b[0], B4b[1]);
        c = __builtin_amdgcn_wmma_f32_16x16x32_bf16(false, a.bf, false, b.bf,
                                                    (short)0, c, false, false);
    }

    const int colg   = wave * 16 + r16;
    const float bias = Qb[colg];
    const int rhi    = (lane >> 4) * 8;      // C/D layout: VGPR v -> row v + 8*(lane>=16)
#pragma unroll
    for (int v = 0; v < 8; ++v) {
        float val = c[v] + bias;
        out[(size_t)(mt * 16 + v + rhi) * D_HID + colg] = val > 0.f ? val : 0.f;
    }
}

// ---------------------------------------------------------------------------
// Edge scatter: one wave per edge; each lane handles 4 of 128 features.
// Fire-and-forget relaxed agent-scope atomics -> global_atomic_add_f32.
// ---------------------------------------------------------------------------
__global__ void __launch_bounds__(256)
edge_scatter(const float* __restrict__ nsrc, const float* __restrict__ w,
             const int* __restrict__ esrc, const int* __restrict__ edst,
             float* __restrict__ accum, float* __restrict__ wsum, int E) {
    const int e = blockIdx.x * 8 + (threadIdx.x >> 5);
    if (e >= E) return;
    const int lane = threadIdx.x & 31;
    const int s = esrc[e];
    const int d = edst[e];
    const float we = w[e];

    float4 v = *((const float4*)(nsrc + (size_t)s * D_HID) + lane);
    float* dst = accum + (size_t)d * D_HID + lane * 4;
    __hip_atomic_fetch_add(dst + 0, v.x * we, __ATOMIC_RELAXED, __HIP_MEMORY_SCOPE_AGENT);
    __hip_atomic_fetch_add(dst + 1, v.y * we, __ATOMIC_RELAXED, __HIP_MEMORY_SCOPE_AGENT);
    __hip_atomic_fetch_add(dst + 2, v.z * we, __ATOMIC_RELAXED, __HIP_MEMORY_SCOPE_AGENT);
    __hip_atomic_fetch_add(dst + 3, v.w * we, __ATOMIC_RELAXED, __HIP_MEMORY_SCOPE_AGENT);
    if (lane == 0)
        __hip_atomic_fetch_add(wsum + d, we, __ATOMIC_RELAXED, __HIP_MEMORY_SCOPE_AGENT);
}

// ---------------------------------------------------------------------------
// GEMM2: z[m,n] = relu( inv_m * (accum[m,:] @ B1[:,n]) + h_dst[m,:] @ B2[:,n]
//                       + Wb[n] ),   inv_m = 1/max(wsum[m],1).
// Two f32 accumulators (one per K-half); the 1/ws row scale is applied once in
// the epilogue to the f32 partial sums (cheaper AND more accurate than scaling
// bf16 inputs). K-steps of 32 never straddle the 128 boundary.
// ---------------------------------------------------------------------------
__global__ void __launch_bounds__(256)
gemm2_relu(const float* __restrict__ accum, const float* __restrict__ wsum,
           const float* __restrict__ hdst, const float* __restrict__ Ww,
           const float* __restrict__ Wb, float* __restrict__ out) {
    const int wave  = threadIdx.x >> 5;
    const int lane  = threadIdx.x & 31;
    const int mt    = blockIdx.x;
    const int r16   = lane & 15;
    const int khalf = lane >> 4;

    const int mg = mt * 16 + r16;
    const float* Nrow = accum + (size_t)mg * D_HID;
    const float* Hrow = hdst  + (size_t)mg * D_IN;
    const float* Brow = Ww + (size_t)(wave * 16 + r16) * D_K2;

    v8f c1 = {};   // accum half  (K = 0..127)
    v8f c2 = {};   // h_dst half  (K = 128..255)
#pragma unroll
    for (int k0 = 0; k0 < D_K2; k0 += 32) {
        const float* src = (k0 < D_HID) ? (Nrow + k0) : (Hrow + k0 - D_HID);
        const float4* A4a = (const float4*)(src + khalf * 8);
        const float4* A4b = (const float4*)(src + 16 + khalf * 8);
        const float4* B4a = (const float4*)(Brow + k0 + khalf * 8);
        const float4* B4b = (const float4*)(Brow + k0 + 16 + khalf * 8);
        V16 a, b;
        pack8(a, 0, A4a[0], A4a[1]);
        pack8(a, 4, A4b[0], A4b[1]);
        pack8(b, 0, B4a[0], B4a[1]);
        pack8(b, 4, B4b[0], B4b[1]);
        if (k0 < D_HID)
            c1 = __builtin_amdgcn_wmma_f32_16x16x32_bf16(false, a.bf, false, b.bf,
                                                         (short)0, c1, false, false);
        else
            c2 = __builtin_amdgcn_wmma_f32_16x16x32_bf16(false, a.bf, false, b.bf,
                                                         (short)0, c2, false, false);
    }

    const int colg   = wave * 16 + r16;
    const float bias = Wb[colg];
    const int rhi    = (lane >> 4) * 8;      // C/D layout: VGPR v -> row v + 8*(lane>=16)
#pragma unroll
    for (int v = 0; v < 8; ++v) {
        const int row = mt * 16 + v + rhi;
        const float invw = 1.0f / fmaxf(wsum[row], 1.0f);
        float val = c1[v] * invw + c2[v] + bias;
        out[(size_t)row * D_HID + colg] = val > 0.f ? val : 0.f;
    }
}

// ---------------------------------------------------------------------------
extern "C" void kernel_launch(void* const* d_in, const int* in_sizes, int n_in,
                              void* d_out, int out_size, void* d_ws, size_t ws_size,
                              hipStream_t stream) {
    const float* h_src    = (const float*)d_in[0];
    const float* h_dst    = (const float*)d_in[1];
    const float* weights  = (const float*)d_in[2];
    const int*   edge_src = (const int*)d_in[3];
    const int*   edge_dst = (const int*)d_in[4];
    const float* Q_w      = (const float*)d_in[5];
    const float* Q_b      = (const float*)d_in[6];
    const float* W_w      = (const float*)d_in[7];
    const float* W_b      = (const float*)d_in[8];
    float*       out      = (float*)d_out;
    const int E = in_sizes[2];

    // Workspace layout: n_src | accum | wsum   (all f32)
    float* nsrc  = (float*)d_ws;
    float* accum = nsrc  + (size_t)N_SRC * D_HID;
    float* wsum  = accum + (size_t)N_DST * D_HID;

    // 1) zero accum + wsum (contiguous region)
    {
        long n4 = ((long)N_DST * D_HID + N_DST) / 4;
        zero_f4<<<2048, 256, 0, stream>>>((float4*)accum, n4);
    }
    // 2) n_src = relu(h_src @ Q_w^T + Q_b)
    gemm1_relu<<<N_SRC / 16, 256, 0, stream>>>(h_src, Q_w, Q_b, nsrc);
    // 3) weighted scatter-add over edges
    edge_scatter<<<(E + 7) / 8, 256, 0, stream>>>(nsrc, weights, edge_src, edge_dst,
                                                  accum, wsum, E);
    // 4) z = relu(concat(accum/max(wsum,1), h_dst) @ W_w^T + W_b)
    gemm2_relu<<<N_DST / 16, 256, 0, stream>>>(accum, wsum, h_dst, W_w, W_b, out);
}
